// Group_10_35811437314234
// MI455X (gfx1250) — matrix operand order repaired and checked by reading, not compiled, not run on hardware
//
#include <hip/hip_runtime.h>

// CDNA5 / gfx1250: implicit-GEMM 3x3 conv (replicate pad) + phase-shuffle epilogue.
// GEMM: M=9728 (Cout) x N=4096 (B*H*W) x K=4608, computed as bf16x3 split
// (hi+lo Dekker) on v_wmma_f32_16x16x32_bf16, f32 accumulation.
// K is ordered kernel-position-major (k = r*512 + ci) so both the weight and
// im2col gathers become constant-stride loads with immediate offsets.

typedef __attribute__((ext_vector_type(16))) __bf16 v16bf;
typedef __attribute__((ext_vector_type(8)))  __bf16 v8bf;
typedef __attribute__((ext_vector_type(8)))  float  v8f;

#define TILE_PITCH 40   // bf16 elems per LDS row: 32 data + 8 pad (16B-aligned chunks)

static __device__ inline v16bf cat16(v8bf a, v8bf b) {
  return __builtin_shufflevector(a, b, 0,1,2,3,4,5,6,7,8,9,10,11,12,13,14,15);
}

__global__ __launch_bounds__(256, 1) void conv3x3_ps_wmma(
    const float* __restrict__ X,     // [4,512,32,32]
    const float* __restrict__ Wm,    // [9728,512,3,3] == [9728,4608] row-major
    const float* __restrict__ bias,  // [9728]
    float* __restrict__ out)         // [4,152,256,256] (phase-shuffled)
{
  __shared__ __bf16 Ahi[128 * TILE_PITCH];
  __shared__ __bf16 Alo[128 * TILE_PITCH];
  __shared__ __bf16 Bhi[128 * TILE_PITCH];
  __shared__ __bf16 Blo[128 * TILE_PITCH];

  const int t     = threadIdx.x;
  const int lane  = t & 31;
  const int wid   = t >> 5;       // 8 wave32s
  const int waveM = wid >> 1;     // 0..3 -> 32 rows each
  const int waveN = wid & 1;      // 0..1 -> 64 cols each

  const int n0 = blockIdx.x * 128;   // pixel-column tile (32 tiles)
  const int m0 = blockIdx.y * 128;   // Cout tile (76 tiles)

  // ---- staging roles: 2 threads per row, 16 K (ci) values each ----
  const int rowS = t >> 1;          // 0..127
  const int kS   = (t & 1) << 4;    // 0 or 16

  const int pS = n0 + rowS;                 // pixel index = b*1024 + h*32 + w
  const int bS = pS >> 10;
  const int hS = (pS >> 5) & 31;
  const int wS = pS & 31;
  // replicate-pad clamped row/col offsets (precomputed once)
  const int hrow3[3] = { (hS > 0 ? hS - 1 : 0) * 32, hS * 32, (hS < 31 ? hS + 1 : 31) * 32 };
  const int wcol3[3] = { (wS > 0 ? wS - 1 : 0),       wS,      (wS < 31 ? wS + 1 : 31)      };

  const float* __restrict__ Xb = X + (size_t)bS * (512 * 1024);

  v8f acc[2][4] = {};

  const int mrow = lane & 15;
  const int kkA  = (lane >> 4) << 3;    // A fragment K-chunk select {0,8}
  const int kkB  = (lane >> 4) << 3;    // B fragment K-chunk select {0,8}

  for (int r9 = 0; r9 < 9; ++r9) {      // kernel position r = kh*3+kw
    const int kh = r9 / 3;
    const int kw = r9 - 3 * kh;
    const int hrow = (kh == 0) ? hrow3[0] : ((kh == 1) ? hrow3[1] : hrow3[2]);
    const int wcol = (kw == 0) ? wcol3[0] : ((kw == 1) ? wcol3[1] : wcol3[2]);

    // per-r thread base pointers
    const float* __restrict__ Bbase = Xb + (size_t)kS * 1024 + hrow + wcol;
    const float* __restrict__ Abase =
        Wm + (size_t)(m0 + rowS) * 4608 + (size_t)kS * 9 + r9;

    for (int c16 = 0; c16 < 16; ++c16) { // ci chunk of 32 (this thread: 16)
      const int cc = c16 * 32;

      // ---- global: A = W[m][cc+kS+i][r9], stride 36B, immediate offsets ----
      float a[16];
      {
        const float* __restrict__ Ap = Abase + (size_t)cc * 9;
        #pragma unroll
        for (int i = 0; i < 16; ++i) a[i] = Ap[i * 9];
        __builtin_prefetch(Ap + 32 * 9, 0, 0);     // next ci-chunk
      }
      // ---- global: B = x[b][cc+kS+i][hh][ww], stride 4096B, imm offsets ----
      float bv[16];
      {
        const float* __restrict__ Bp = Bbase + (size_t)cc * 1024;
        #pragma unroll
        for (int i = 0; i < 16; ++i) bv[i] = Bp[i * 1024];
        __builtin_prefetch(Bp + 32 * 1024, 0, 0);  // next ci-chunk
      }

      __syncthreads();   // prior iteration's fragment reads complete

      // ---- split f32 -> bf16 hi + lo (Dekker), store to LDS b128 chunks ----
      {
        alignas(16) __bf16 ah[16], al[16], bh[16], bl[16];
        #pragma unroll
        for (int i = 0; i < 16; ++i) {
          __bf16 h = (__bf16)a[i];
          ah[i] = h;  al[i] = (__bf16)(a[i] - (float)h);
          __bf16 g = (__bf16)bv[i];
          bh[i] = g;  bl[i] = (__bf16)(bv[i] - (float)g);
        }
        const int o = rowS * TILE_PITCH + kS;
        *(v8bf*)&Ahi[o]     = *(const v8bf*)&ah[0];
        *(v8bf*)&Ahi[o + 8] = *(const v8bf*)&ah[8];
        *(v8bf*)&Alo[o]     = *(const v8bf*)&al[0];
        *(v8bf*)&Alo[o + 8] = *(const v8bf*)&al[8];
        *(v8bf*)&Bhi[o]     = *(const v8bf*)&bh[0];
        *(v8bf*)&Bhi[o + 8] = *(const v8bf*)&bh[8];
        *(v8bf*)&Blo[o]     = *(const v8bf*)&bl[0];
        *(v8bf*)&Blo[o + 8] = *(const v8bf*)&bl[8];
      }

      __syncthreads();

      // ---- fragments (ISA layouts):
      // A 16x32 bf16: lane m=L%16; lanes<16 -> K {0..7,16..23}; lanes>=16 -> +8
      // B 32x16 bf16: lane n=L%16; lanes<16 -> K {0..7,16..23}; lanes>=16 -> +8
      v16bf afh[2], afl[2], bfh[4], bfl[4];
      #pragma unroll
      for (int tm = 0; tm < 2; ++tm) {
        const int o = (waveM * 32 + tm * 16 + mrow) * TILE_PITCH;
        afh[tm] = cat16(*(const v8bf*)&Ahi[o + kkA], *(const v8bf*)&Ahi[o + kkA + 16]);
        afl[tm] = cat16(*(const v8bf*)&Alo[o + kkA], *(const v8bf*)&Alo[o + kkA + 16]);
      }
      #pragma unroll
      for (int tn = 0; tn < 4; ++tn) {
        const int o = (waveN * 64 + tn * 16 + mrow) * TILE_PITCH + kkB;
        bfh[tn] = cat16(*(const v8bf*)&Bhi[o], *(const v8bf*)&Bhi[o + 16]);
        bfl[tn] = cat16(*(const v8bf*)&Blo[o], *(const v8bf*)&Blo[o + 16]);
      }

      // ---- 24 WMMAs: hi*hi + lo*hi + hi*lo (drop lo*lo) ----
      #pragma unroll
      for (int tm = 0; tm < 2; ++tm)
        #pragma unroll
        for (int tn = 0; tn < 4; ++tn) {
          acc[tm][tn] = __builtin_amdgcn_wmma_f32_16x16x32_bf16(
              false, afh[tm], false, bfh[tn], (short)0, acc[tm][tn], false, false);
          acc[tm][tn] = __builtin_amdgcn_wmma_f32_16x16x32_bf16(
              false, afl[tm], false, bfh[tn], (short)0, acc[tm][tn], false, false);
          acc[tm][tn] = __builtin_amdgcn_wmma_f32_16x16x32_bf16(
              false, afh[tm], false, bfl[tn], (short)0, acc[tm][tn], false, false);
        }
    }
  }

  // ---- epilogue: bias + phase-shuffle scatter (non-temporal: keep W in L2) ----
  // C/D layout: VGPR r, lanes<16 -> (M=r, N=lane); lanes>=16 -> (M=r+8, N=lane-16)
  const int laneHalf = (lane >> 4) << 3;
  const int ncol     = lane & 15;
  #pragma unroll
  for (int tm = 0; tm < 2; ++tm)
    #pragma unroll
    for (int tn = 0; tn < 4; ++tn)
      #pragma unroll
      for (int r = 0; r < 8; ++r) {
        const int co = m0 + waveM * 32 + tm * 16 + laneHalf + r;
        const int p  = n0 + waveN * 64 + tn * 16 + ncol;
        const int bI = p >> 10, hI = (p >> 5) & 31, wI = p & 31;
        const int c6 = co & 63, nc = co >> 6;
        const int u  = ((c6 >> 1) << 3) + (wI & 7);
        const int v  = ((((c6 & 1) << 4) + (hI >> 1)) << 3) + ((hI & 1) << 2) + (wI >> 3);
        const size_t oi = (((size_t)bI * 152 + nc) * 256 + u) * 256 + v;
        __builtin_nontemporal_store(acc[tm][tn][r] + bias[co], &out[oi]);
      }
}

extern "C" void kernel_launch(void* const* d_in, const int* in_sizes, int n_in,
                              void* d_out, int out_size, void* d_ws, size_t ws_size,
                              hipStream_t stream) {
  const float* x    = (const float*)d_in[0];   // [4,512,32,32]
  const float* Wm   = (const float*)d_in[1];   // [9728,512,3,3]
  const float* bias = (const float*)d_in[2];   // [9728]
  float* out = (float*)d_out;                  // [4,152,256,256]

  dim3 grid(32, 76);   // N tiles (4096/128) x M tiles (9728/128)
  conv3x3_ps_wmma<<<grid, 256, 0, stream>>>(x, Wm, bias, out);
}
